// Model_11733850653045
// MI455X (gfx1250) — compile-verified, matrix operand order checked by reading, not run on hardware
//
#include <hip/hip_runtime.h>

// ---------------------------------------------------------------------------
// EA-LSTM on MI455X (gfx1250, wave32).
//   kernel 1: pre-swizzle [w_hh; w_ih; bias] -> bf16 WMMA B-fragments in ws
//   kernel 2: persistent recurrent kernel, 32 WGs x 16 batch rows, T=365 steps
//             gates = [h | x_t | 1] @ Wfrag  via v_wmma_f32_16x16x32_bf16
//             B streams from L2 as global_load_b128 (hoisting blocked),
//             A preloaded from LDS, native v_tanh for gate math.
//   kernel 3: fc head (dot(h_T, fc_w) + fc_b) with wave shuffle reduction
// ---------------------------------------------------------------------------

#define NB   512     // batch
#define TT   365     // timesteps
#define DDYN 16
#define DSTA 32
#define HH   256
#define G3   768     // 3*H
#define KT   9       // K tiles of 32: 256 (h) + 16 (x) + 1 (bias one) + pad
#define NT   48      // N tiles of 16 (768/16)
#define BM   16      // batch rows per workgroup
#define THREADS 512  // 16 waves
#define GST  772     // gates LDS row stride (floats); ≡4 mod 64 banks

#define GLOBAL_AS __attribute__((address_space(1)))

typedef __attribute__((ext_vector_type(16))) __bf16 v16bf;
typedef __attribute__((ext_vector_type(8)))  float  v8f;
typedef __attribute__((ext_vector_type(4)))  float  v4f;
typedef __attribute__((ext_vector_type(2)))  __bf16 v2bf;

__device__ __forceinline__ unsigned short f2bf(float f) {
    unsigned int u = __float_as_uint(f);
    u += 0x7FFFu + ((u >> 16) & 1u);          // round to nearest even
    return (unsigned short)(u >> 16);
}
__device__ __forceinline__ unsigned int pack2bf(float a, float b) {
#if __has_builtin(__builtin_amdgcn_cvt_pk_bf16_f32)
    v2bf p = __builtin_amdgcn_cvt_pk_bf16_f32(a, b);
    return __builtin_bit_cast(unsigned int, p);
#else
    return (unsigned int)f2bf(a) | ((unsigned int)f2bf(b) << 16);
#endif
}
__device__ __forceinline__ float fast_tanh(float x) {
#if __has_builtin(__builtin_amdgcn_tanhf)
    return __builtin_amdgcn_tanhf(x);
#elif __has_builtin(__builtin_amdgcn_tanh_f32)
    return __builtin_amdgcn_tanh_f32(x);
#else
    float a = __builtin_fabsf(x);
    float e = __expf(-2.0f * a);
    float t = (1.0f - e) / (1.0f + e);
    return __builtin_copysignf(t, x);
#endif
}
__device__ __forceinline__ float fast_sigmoid(float x) {
    // sigmoid(x) = 0.5 + 0.5*tanh(x/2): one TRANS op + FMAs on gfx1250
    return __builtin_fmaf(0.5f, fast_tanh(0.5f * x), 0.5f);
}

// --------------------------------------------------------------------------
// Kernel 1: build bf16 B-fragments for WMMA from [w_hh(256) ; w_ih(16) ;
// bias(1) ; zero-pad(15)] x 768.  Fragment layout (per 32x16 K x N tile):
//   lane = n_local + 16*hi ; halves: k_in_tile = (half&7) + 8*hi + 16*(half>>3)
// --------------------------------------------------------------------------
__global__ void wfrag_build(const float* __restrict__ w_ih,
                            const float* __restrict__ w_hh,
                            const float* __restrict__ bias,
                            unsigned short* __restrict__ wf) {
    int idx = blockIdx.x * blockDim.x + threadIdx.x;
    if (idx >= KT * NT * 32 * 16) return;
    int half = idx & 15;
    int lane = (idx >> 4) & 31;
    int rest = idx >> 9;
    int nt = rest % NT;
    int kt = rest / NT;
    int hi = lane >> 4;
    int nl = lane & 15;
    int kin = (half & 7) + 8 * hi + 16 * (half >> 3);
    int kg = kt * 32 + kin;
    int n  = nt * 16 + nl;
    float v = 0.0f;
    if (kg < HH)               v = w_hh[kg * G3 + n];
    else if (kg < HH + DDYN)   v = w_ih[(kg - HH) * G3 + n];
    else if (kg == HH + DDYN)  v = bias[n];
    wf[idx] = f2bf(v);
}

// --------------------------------------------------------------------------
// Kernel 2: persistent recurrent kernel. grid = NB/BM = 32, block = 512.
// --------------------------------------------------------------------------
__global__ __launch_bounds__(THREADS)
void ealstm_main(const float* __restrict__ x_d, const float* __restrict__ x_s,
                 const float* __restrict__ w_sh, const float* __restrict__ bias_s,
                 const unsigned short* wfrag,
                 float* __restrict__ h_out, float* __restrict__ c_out) {
    __shared__ __align__(16) float          s_gates[BM * GST];     // 49.4 KB
    __shared__ __align__(16) unsigned short s_h[KT * 32 * 16];     //  9.2 KB

    const int tid  = threadIdx.x;
    const int lane = tid & 31;
    const int wid  = tid >> 5;
    const int row0 = blockIdx.x * BM;

    // ---- init: zero A-fragment buffer ----
    for (int i = tid; i < KT * 32 * 16; i += THREADS) s_h[i] = 0;
    __syncthreads();
    // constant-1 column (k_global == 272 -> kt=8, k'=16 -> lane=m, half=8)
    if (tid < BM) s_h[(8 * 32 + tid) * 16 + 8] = 0x3F80;
    // stage x_{t=0} into kt=8 fragment slot (k' 0..15)
    if (tid < BM * DDYN) {
        int m = tid >> 4, k = tid & 15;
        float v = x_d[((size_t)(row0 + m) * TT + 0) * DDYN + k];
        s_h[(8 * 32 + m + ((k >> 3) & 1) * 16) * 16 + (k & 7)] = f2bf(v);
    }

    // ---- per-thread elementwise ownership: row r, two groups of 4 columns
    //      at ca = lane*4 and cb = 128 + lane*4  (conflict-free LDS banks)
    const int r  = tid >> 5;       // 0..15
    const int ca = lane * 4;       // 0..124
    const int cb = 128 + ca;       // 128..252
    float c_reg[8], ig[8];
    #pragma unroll
    for (int e = 0; e < 8; ++e) {
        int col = (e < 4) ? (ca + e) : (cb + e - 4);
        float a = bias_s[col];
        for (int d = 0; d < DSTA; ++d)
            a += x_s[(row0 + r) * DSTA + d] * w_sh[d * HH + col];
        ig[e]    = fast_sigmoid(a);   // static EA input gate
        c_reg[e] = 0.0f;
    }
    __syncthreads();

    const int nt0 = wid * 3;     // 3 N-tiles per wave, 16 waves x 3 = 48

    for (int t = 0; t < TT; ++t) {
        // Launder the weight pointer every iteration (blocks LICM from
        // hoisting 27 B-fragments into registers -> scratch spills), then
        // cast back to the GLOBAL address space so the loads emit as
        // global_load_b128 (LOADcnt only) instead of flat (LOADcnt+DScnt).
        const unsigned short* wfp = wfrag;
        asm volatile("" : "+s"(wfp));
        const GLOBAL_AS v16bf* wfv = (const GLOBAL_AS v16bf*)wfp;

        // ---- preload all 9 A-fragments (batched ds_load_b128s) ----
        v16bf a[KT];
        #pragma unroll
        for (int kt = 0; kt < KT; ++kt)
            a[kt] = *reinterpret_cast<const v16bf*>(&s_h[(kt * 32 + lane) * 16]);

        // ================= GEMM phase: gates = [h|x|1] @ W =================
        v8f acc0 = {}, acc1 = {}, acc2 = {};
        #pragma unroll
        for (int kt = 0; kt < KT; ++kt) {
            v16bf b0 = wfv[(kt * NT + nt0 + 0) * 32 + lane];
            v16bf b1 = wfv[(kt * NT + nt0 + 1) * 32 + lane];
            v16bf b2 = wfv[(kt * NT + nt0 + 2) * 32 + lane];
            acc0 = __builtin_amdgcn_wmma_f32_16x16x32_bf16(false, a[kt], false, b0,
                                                           (short)0, acc0, false, false);
            acc1 = __builtin_amdgcn_wmma_f32_16x16x32_bf16(false, a[kt], false, b1,
                                                           (short)0, acc1, false, false);
            acc2 = __builtin_amdgcn_wmma_f32_16x16x32_bf16(false, a[kt], false, b2,
                                                           (short)0, acc2, false, false);
        }
        // spill accumulators to LDS gates (C/D layout: vgpr v -> M=v+8*(lane>=16))
        {
            int rr = 8 * (lane >> 4);
            int cc = lane & 15;
            #pragma unroll
            for (int v = 0; v < 8; ++v) {
                s_gates[(rr + v) * GST + (nt0 + 0) * 16 + cc] = acc0[v];
                s_gates[(rr + v) * GST + (nt0 + 1) * 16 + cc] = acc1[v];
                s_gates[(rr + v) * GST + (nt0 + 2) * 16 + cc] = acc2[v];
            }
        }
        __syncthreads();

        // ================= elementwise LSTM update =========================
        float fv[8] __attribute__((aligned(16)));
        float ov[8] __attribute__((aligned(16)));
        float gv[8] __attribute__((aligned(16)));
        float hv[8] __attribute__((aligned(16)));
        float cv[8] __attribute__((aligned(16)));
        *(v4f*)&fv[0] = *(const v4f*)&s_gates[r * GST + 0   + ca];
        *(v4f*)&fv[4] = *(const v4f*)&s_gates[r * GST + 0   + cb];
        *(v4f*)&ov[0] = *(const v4f*)&s_gates[r * GST + 256 + ca];
        *(v4f*)&ov[4] = *(const v4f*)&s_gates[r * GST + 256 + cb];
        *(v4f*)&gv[0] = *(const v4f*)&s_gates[r * GST + 512 + ca];
        *(v4f*)&gv[4] = *(const v4f*)&s_gates[r * GST + 512 + cb];
        #pragma unroll
        for (int e = 0; e < 8; ++e) {
            float c  = fast_sigmoid(fv[e]) * c_reg[e] + ig[e] * fast_tanh(gv[e]);
            c_reg[e] = c;
            cv[e]    = c;
            hv[e]    = fast_sigmoid(ov[e]) * fast_tanh(c);
        }
        // stream h_n / c_n (coalesced: wave covers 256 contiguous floats,
        // non-temporal: write-once data must not evict L2-resident weights)
        {
            size_t base = ((size_t)(row0 + r) * TT + t) * HH;
            __builtin_nontemporal_store(*(const v4f*)&hv[0], (v4f*)&h_out[base + ca]);
            __builtin_nontemporal_store(*(const v4f*)&hv[4], (v4f*)&h_out[base + cb]);
            __builtin_nontemporal_store(*(const v4f*)&cv[0], (v4f*)&c_out[base + ca]);
            __builtin_nontemporal_store(*(const v4f*)&cv[4], (v4f*)&c_out[base + cb]);
        }
        // repack h into A-fragment layout for next step (2x ds_store_b64)
        {
            unsigned int pkA[2] __attribute__((aligned(8)));
            unsigned int pkB[2] __attribute__((aligned(8)));
            pkA[0] = pack2bf(hv[0], hv[1]);
            pkA[1] = pack2bf(hv[2], hv[3]);
            pkB[0] = pack2bf(hv[4], hv[5]);
            pkB[1] = pack2bf(hv[6], hv[7]);
            // group A: columns ca..ca+3
            {
                int kt = ca >> 5, k0 = ca & 31;
                int lane_t = r + ((k0 >> 3) & 1) * 16;
                int hb = (k0 & 7) + ((k0 >> 4) << 3);
                *(uint2*)&s_h[(kt * 32 + lane_t) * 16 + hb] = *(const uint2*)&pkA[0];
            }
            // group B: columns cb..cb+3
            {
                int kt = cb >> 5, k0 = cb & 31;
                int lane_t = r + ((k0 >> 3) & 1) * 16;
                int hb = (k0 & 7) + ((k0 >> 4) << 3);
                *(uint2*)&s_h[(kt * 32 + lane_t) * 16 + hb] = *(const uint2*)&pkB[0];
            }
        }
        // stage x_{t+1}
        if (t + 1 < TT && tid < BM * DDYN) {
            int m = tid >> 4, k = tid & 15;
            float v = x_d[((size_t)(row0 + m) * TT + (t + 1)) * DDYN + k];
            s_h[(8 * 32 + m + ((k >> 3) & 1) * 16) * 16 + (k & 7)] = f2bf(v);
        }
        __syncthreads();
    }
}

// --------------------------------------------------------------------------
// Kernel 3: out[b] = h_n[b, T-1, :] . fc_w + fc_b   (one wave per row)
// --------------------------------------------------------------------------
__global__ void fc_kernel(const float* __restrict__ h_n,
                          const float* __restrict__ fc_w,
                          const float* __restrict__ fc_b,
                          float* __restrict__ out) {
    int w    = (blockIdx.x * blockDim.x + threadIdx.x) >> 5;
    int lane = threadIdx.x & 31;
    if (w >= NB) return;
    const float* hrow = h_n + ((size_t)w * TT + (TT - 1)) * HH;
    float s = 0.0f;
    for (int i = lane; i < HH; i += 32) s += hrow[i] * fc_w[i];
    for (int off = 16; off > 0; off >>= 1) s += __shfl_down(s, off, 32);
    if (lane == 0) out[w] = s + fc_b[0];
}

// --------------------------------------------------------------------------
extern "C" void kernel_launch(void* const* d_in, const int* in_sizes, int n_in,
                              void* d_out, int out_size, void* d_ws, size_t ws_size,
                              hipStream_t stream) {
    const float* x_d    = (const float*)d_in[0];
    const float* x_s    = (const float*)d_in[1];
    const float* w_ih   = (const float*)d_in[2];
    const float* w_hh   = (const float*)d_in[3];
    const float* w_sh   = (const float*)d_in[4];
    const float* bias   = (const float*)d_in[5];
    const float* bias_s = (const float*)d_in[6];
    const float* fc_w   = (const float*)d_in[7];
    const float* fc_b   = (const float*)d_in[8];

    float* out = (float*)d_out;                   // [B,1]
    float* h_n = out + NB;                        // [B,T,H]
    float* c_n = h_n + (size_t)NB * TT * HH;      // [B,T,H]

    unsigned short* wf = (unsigned short*)d_ws;   // 432 KB of bf16 fragments

    int total = KT * NT * 32 * 16;
    wfrag_build<<<(total + 255) / 256, 256, 0, stream>>>(w_ih, w_hh, bias, wf);
    ealstm_main<<<NB / BM, THREADS, 0, stream>>>(x_d, x_s, w_sh, bias_s, wf,
                                                 h_n, c_n);
    fc_kernel<<<(NB * 32 + 255) / 256, 256, 0, stream>>>(h_n, fc_w, fc_b, out);
}
